// DConv_9612136809141
// MI455X (gfx1250) — matrix operand order repaired and checked by reading, not compiled, not run on hardware
//
#include <hip/hip_runtime.h>
#include <hip/hip_bf16.h>

typedef __attribute__((ext_vector_type(16))) _Float16 v16h;
typedef __attribute__((ext_vector_type(8)))  float    v8f;
typedef __attribute__((ext_vector_type(8)))  unsigned int v8u;

#define BATCH 8
#define CCH   128
#define HH_   64
#define WW_   64
#define PIX   (HH_ * WW_)          // 4096
#define KK    9
#define POUT  128
#define KTOT  (CCH * KK)           // 1152
#define NKSTEP (KTOT / 32)         // 36 (even)
#define ASTRIDE 1160               // 1152 + 8 pad halves; 2*1160 = 2320 = 16*145 (16B aligned rows)

// ---------------- Kernel 0: pack deform weights into WMMA B-fragment layout (f16) ----------------
// packed index: ((ntile*36 + kstep)*32 + lane)*8 + v  -> dword holding (B[k][n], B[k+1][n])
// k = kstep*32 + ((lane&16)?8:0) + 2*v + ((v&4)?8:0) ; n = ntile*16 + (lane&15) ; K = kpos*128 + c
__global__ __launch_bounds__(256) void pack_weights(const float* __restrict__ dwgt,
                                                    unsigned int* __restrict__ pb) {
    int idx = blockIdx.x * 256 + threadIdx.x;
    if (idx >= 8 * NKSTEP * 32 * 8) return;
    int v     = idx & 7;
    int lane  = (idx >> 3) & 31;
    int rest  = idx >> 8;
    int kstep = rest % NKSTEP;
    int ntile = rest / NKSTEP;
    int n     = ntile * 16 + (lane & 15);
    int kbase = kstep * 32 + ((lane & 16) ? 8 : 0) + 2 * v + ((v & 4) ? 8 : 0);
    unsigned int packed = 0;
    #pragma unroll
    for (int h = 0; h < 2; ++h) {
        int k    = kbase + h;
        int c    = k & 127;
        int kpos = k >> 7;
        float w  = dwgt[n * KTOT + c * KK + kpos];
        _Float16 hf = (_Float16)w;
        packed |= ((unsigned int)__builtin_bit_cast(unsigned short, hf)) << (16 * h);
    }
    pb[idx] = packed;
}

// ---------------- Kernel 1: depthwise 3x3 conv + bias + BN + ReLU -> h ----------------
__global__ __launch_bounds__(256) void dw_bn_relu(const float* __restrict__ x,
                                                  const float* __restrict__ w,
                                                  const float* __restrict__ bias,
                                                  const float* __restrict__ gamma,
                                                  const float* __restrict__ beta,
                                                  const float* __restrict__ mean,
                                                  const float* __restrict__ var,
                                                  float* __restrict__ h) {
    int idx = blockIdx.x * 256 + threadIdx.x;
    if (idx >= BATCH * CCH * PIX) return;
    int wwp = idx & 63;
    int hhp = (idx >> 6) & 63;
    int c   = (idx >> 12) & 127;
    int b   = idx >> 19;
    const float* xp = x + (((long)(b * CCH + c)) << 12);
    const float* wp = w + c * 9;
    float acc = bias[c];
    #pragma unroll
    for (int ky = 0; ky < 3; ++ky) {
        int y = hhp + ky - 1;
        if ((unsigned)y >= (unsigned)HH_) continue;
        #pragma unroll
        for (int kx = 0; kx < 3; ++kx) {
            int xx = wwp + kx - 1;
            if ((unsigned)xx >= (unsigned)WW_) continue;
            acc = fmaf(xp[(y << 6) + xx], wp[ky * 3 + kx], acc);
        }
    }
    float inv = rsqrtf(var[c] + 1e-5f);
    float r = (acc - mean[c]) * (gamma[c] * inv) + beta[c];
    h[idx] = fmaxf(r, 0.0f);
}

// ---------------- Kernel 2: 1x1 offset conv (128 -> 18) ----------------
__global__ __launch_bounds__(256) void offset_conv(const float* __restrict__ h,
                                                   const float* __restrict__ ow,
                                                   float* __restrict__ off) {
    int idx = blockIdx.x * 256 + threadIdx.x;
    if (idx >= BATCH * 18 * PIX) return;
    int pix = idx & 4095;
    int o   = (idx >> 12) % 18;
    int b   = idx / (18 * PIX);
    const float* hp = h + (((long)b * CCH) << 12) + pix;
    const float* wp = ow + o * CCH;
    float acc = 0.0f;
    #pragma unroll 4
    for (int c = 0; c < CCH; ++c)
        acc = fmaf(hp[(long)c << 12], wp[c], acc);
    off[idx] = acc;
}

// ---------------- Kernel 3: deformable conv via WMMA ----------------
__device__ inline float samp(const float* __restrict__ xc, int y, int xx) {
    if ((unsigned)y < (unsigned)HH_ && (unsigned)xx < (unsigned)WW_)
        return xc[(y << 6) + xx];
    return 0.0f;
}

__device__ inline v8u mk8(uint4 lo, uint4 hi) {
    v8u r;
    r[0] = lo.x; r[1] = lo.y; r[2] = lo.z; r[3] = lo.w;
    r[4] = hi.x; r[5] = hi.y; r[6] = hi.z; r[7] = hi.w;
    return r;
}

__global__ __launch_bounds__(256) void deform_wmma(const float* __restrict__ x,
                                                   const float* __restrict__ off,
                                                   const unsigned int* __restrict__ pb,
                                                   float* __restrict__ out) {
    __shared__ unsigned short sA[16 * ASTRIDE];   // 16 pixels x 1152 K values (f16), padded
    __shared__ int   sY0[144], sX0[144];
    __shared__ float sWY[144], sWX[144];

    int tid      = threadIdx.x;
    int pix_base = blockIdx.x * 16;               // tile of 16 output pixels (same b, same row)
    int b        = pix_base >> 12;
    int pbase    = pix_base & 4095;
    int hh       = pbase >> 6;
    int w0       = pbase & 63;

    // ---- Stage 1: per-(m,kpos) bilinear params ----
    if (tid < 144) {
        int m = tid / 9, kpos = tid % 9;
        int ky = kpos / 3, kx = kpos % 3;
        int wwp = w0 + m;
        const float* op = off + (((long)(b * 18 + kpos * 2)) << 12) + (hh << 6) + wwp;
        float dy = op[0];
        float dx = op[PIX];
        float py = (float)(hh - 1 + ky) + dy;
        float px = (float)(wwp - 1 + kx) + dx;
        float fy = floorf(py), fx = floorf(px);
        sY0[tid] = (int)fy;  sX0[tid] = (int)fx;
        sWY[tid] = py - fy;  sWX[tid] = px - fx;
    }
    __syncthreads();

    // ---- Stage 2: bilinear gather -> f16 A-tile in LDS (K = kpos*128 + c) ----
    const float* xb = x + ((long)b << 19);        // b * 128 * 4096
    for (int t = tid; t < 144 * CCH; t += 256) {
        int c  = t & 127;
        int mk = t >> 7;                          // 0..143
        int m = mk / 9, kpos = mk % 9;
        int y0 = sY0[mk], x0 = sX0[mk];
        float wy = sWY[mk], wx = sWX[mk];
        const float* xc = xb + ((long)c << 12);
        float v00 = samp(xc, y0,     x0);
        float v01 = samp(xc, y0,     x0 + 1);
        float v10 = samp(xc, y0 + 1, x0);
        float v11 = samp(xc, y0 + 1, x0 + 1);
        float val = v00 * (1.0f - wy) * (1.0f - wx)
                  + v01 * (1.0f - wy) * wx
                  + v10 * wy * (1.0f - wx)
                  + v11 * wy * wx;
        _Float16 hv = (_Float16)val;
        sA[m * ASTRIDE + kpos * CCH + c] = __builtin_bit_cast(unsigned short, hv);
    }
    __syncthreads();

    // ---- Stage 3: each wave owns one 16-channel N-tile; 36 WMMA steps over K=1152 ----
    // Two independent accumulators (even/odd k-steps) break the WMMA->WMMA RAW chain
    // (f16 WMMA D->C hazard costs up to 5 slots when serialized).
    int wave = tid >> 5;
    int lane = tid & 31;
    int mrow = lane & 15;
    int kb   = (lane & 16) ? 8 : 0;               // A-fragment K sub-base per ISA layout

    const uint4* arow = (const uint4*)(sA + mrow * ASTRIDE);          // 16B-aligned row
    const unsigned int* bp = pb + (((long)(wave * NKSTEP) * 32 + lane) << 3);

    v8f acc0 = {};
    v8f acc1 = {};
    #pragma unroll 2
    for (int ks = 0; ks < NKSTEP; ks += 2) {
        // ---- even k-step -> acc0 ----
        uint4 a0 = arow[4 * ks +     (kb >> 3)];
        uint4 a1 = arow[4 * ks + 2 + (kb >> 3)];
        const uint4* bq0 = (const uint4*)(bp + ks * 256);
        uint4 b0 = bq0[0];
        uint4 b1 = bq0[1];
        // ---- odd k-step -> acc1 ----
        uint4 a2 = arow[4 * (ks + 1) +     (kb >> 3)];
        uint4 a3 = arow[4 * (ks + 1) + 2 + (kb >> 3)];
        const uint4* bq1 = (const uint4*)(bp + (ks + 1) * 256);
        uint4 b2 = bq1[0];
        uint4 b3 = bq1[1];
        __builtin_prefetch(bp + (ks + 2) * 256, 0, 1);

        v16h av0 = __builtin_bit_cast(v16h, mk8(a0, a1));
        v16h bv0 = __builtin_bit_cast(v16h, mk8(b0, b1));
        acc0 = __builtin_amdgcn_wmma_f32_16x16x32_f16(
                   false, av0, false, bv0, (short)0, acc0, false, false);

        v16h av1 = __builtin_bit_cast(v16h, mk8(a2, a3));
        v16h bv1 = __builtin_bit_cast(v16h, mk8(b2, b3));
        acc1 = __builtin_amdgcn_wmma_f32_16x16x32_f16(
                   false, av1, false, bv1, (short)0, acc1, false, false);
    }

    v8f acc = acc0 + acc1;

    // ---- Store: C/D layout -> lane holds N = lane&15, M = r + 8*(lane>=16) ----
    int nloc = lane & 15;
    int moff = (lane & 16) ? 8 : 0;
    int n    = wave * 16 + nloc;
    float* op = out + (((long)(b * POUT + n)) << 12) + pbase + moff;
    float4 o0 = make_float4(acc[0], acc[1], acc[2], acc[3]);
    float4 o1 = make_float4(acc[4], acc[5], acc[6], acc[7]);
    *(float4*)(op)     = o0;
    *(float4*)(op + 4) = o1;
}

// ---------------- Host launch ----------------
extern "C" void kernel_launch(void* const* d_in, const int* in_sizes, int n_in,
                              void* d_out, int out_size, void* d_ws, size_t ws_size,
                              hipStream_t stream) {
    const float* x         = (const float*)d_in[0];
    const float* dw_weight = (const float*)d_in[1];
    const float* dw_bias   = (const float*)d_in[2];
    const float* bn_gamma  = (const float*)d_in[3];
    const float* bn_beta   = (const float*)d_in[4];
    const float* bn_mean   = (const float*)d_in[5];
    const float* bn_var    = (const float*)d_in[6];
    const float* off_w     = (const float*)d_in[7];
    const float* deform_w  = (const float*)d_in[8];
    float* out = (float*)d_out;

    char* ws = (char*)d_ws;
    float*        h_buf   = (float*)ws;                                   // 16,777,216 B
    float*        off_buf = (float*)(ws + 16777216);                      //  2,359,296 B
    unsigned int* pb      = (unsigned int*)(ws + 16777216 + 2359296);     //    294,912 B

    // 0) pack weights: 8*36*32*8 = 73728 dwords
    pack_weights<<<(73728 + 255) / 256, 256, 0, stream>>>(deform_w, pb);

    // 1) depthwise conv + BN + ReLU: 8*128*4096 = 4,194,304 elements
    dw_bn_relu<<<(BATCH * CCH * PIX) / 256, 256, 0, stream>>>(
        x, dw_weight, dw_bias, bn_gamma, bn_beta, bn_mean, bn_var, h_buf);

    // 2) offset conv: 8*18*4096 = 589,824 elements
    offset_conv<<<(BATCH * 18 * PIX) / 256, 256, 0, stream>>>(h_buf, off_w, off_buf);

    // 3) deformable conv via WMMA: 32768 pixels / 16 per block = 2048 blocks
    deform_wmma<<<(BATCH * PIX) / 16, 256, 0, stream>>>(x, off_buf, pb, out);
}
